// LIFNetwork_45131516346601
// MI455X (gfx1250) — matrix operand order repaired and checked by reading, not compiled, not run on hardware
//
#include <hip/hip_runtime.h>
#include <math.h>

#define T_STEPS 512
#define BATCH   512
#define NIN     8
#define H1      256
#define H2      128
#define BT      16          // batch rows per block
#define BETA1   0.95f
#define BETA2   0.9f
#define THRESH  1.0f
#define SPK_STRIDE 280      // halves per LDS row: 560B = 35*16 (16B-aligned chunks, bank-spread)

typedef __attribute__((ext_vector_type(16))) _Float16 v16h;
typedef __attribute__((ext_vector_type(8)))  _Float16 v8h;
typedef __attribute__((ext_vector_type(8)))  float    v8f;
typedef __attribute__((ext_vector_type(4)))  float    v4f;

__global__ __launch_bounds__(256, 1)
void lif_snn_kernel(const float* __restrict__ x,      // (T,B,8)
                    const float* __restrict__ W1,     // (256,8)
                    const float* __restrict__ W2,     // (128,256)
                    const float* __restrict__ Wout,   // (1,128)
                    const float* __restrict__ bout,   // (1,)
                    float* __restrict__ spk_out,      // (T,B,128)
                    float* __restrict__ mem_out,      // (T,B,128)
                    float* __restrict__ prob_out)     // (B,)
{
  __shared__ __align__(16) _Float16 s_spk1[BT][SPK_STRIDE]; // layer-1 spikes, f16
  __shared__ __align__(16) float    s_x[BT][NIN];           // staged input slab
  __shared__ float                  s_mean[BT][H2];         // mean firing for head

  const int tid   = threadIdx.x;
  const int w     = tid >> 5;     // wave id 0..7  -> H2 N-tile
  const int lane  = tid & 31;
  const int lo    = lane & 15;
  const int hi    = lane >> 4;
  const int bBase = blockIdx.x * BT;

  // ---- one-time: W2 B-matrix fragments in f16 registers (K=256 -> 8 chunks of 32)
  // B layout used here (16x16x32 f16, contiguous k, A matches): lane(lo,hi) holds B[K=16*hi+j][N=lo], j=0..15
  // B[k][n] = W2[16w+lo][k]  (cur2 = spk1 @ W2^T), contiguous 16 floats.
  v16h b2f[8];
  {
    const float* w2row = W2 + (size_t)(16 * w + lo) * H1;
    #pragma unroll
    for (int c = 0; c < 8; ++c) {
      const float* src = w2row + 32 * c + 16 * hi;
      #pragma unroll
      for (int j = 0; j < 16; ++j) b2f[c][j] = (_Float16)src[j];
    }
  }
  // ---- one-time: W1 rows for this lane's two H1 columns (16*(2w+t1)+lo)
  float w1r[2][NIN];
  #pragma unroll
  for (int t1 = 0; t1 < 2; ++t1) {
    const float* w1row = W1 + (size_t)(16 * (2 * w + t1) + lo) * NIN;
    #pragma unroll
    for (int j = 0; j < NIN; ++j) w1r[t1][j] = w1row[j];
  }

  // persistent LIF state in C-layout registers: element (m = r+8*hi, n = lo)
  float mem1[2][8];
  v8f   mem2;
  float spkSum[8];
  #pragma unroll
  for (int r = 0; r < 8; ++r) {
    mem1[0][r] = 0.f; mem1[1][r] = 0.f; mem2[r] = 0.f; spkSum[r] = 0.f;
  }

  for (int t = 0; t < T_STEPS; ++t) {
    // ---- phase 1: stage x_t (16 rows x 8 floats, coalesced 512B, read-once -> NT)
    if (tid < BT * NIN) {
      const int row = tid >> 3, col = tid & 7;
      s_x[row][col] = __builtin_nontemporal_load(
          &x[(size_t)t * BATCH * NIN + (size_t)(bBase + row) * NIN + col]);
    }
    __syncthreads();   // x ready; also fences prev-step spk1 readers vs writers below

    // ---- phase 2: layer-1 LIF (K=8 GEMM on VALU) + f16 spike write to LDS
    #pragma unroll
    for (int r = 0; r < 8; ++r) {
      const int m = r + 8 * hi;                       // batch row in tile
      v4f xa = *(const v4f*)&s_x[m][0];
      v4f xb = *(const v4f*)&s_x[m][4];
      #pragma unroll
      for (int t1 = 0; t1 < 2; ++t1) {
        float cur = xa[0]*w1r[t1][0] + xa[1]*w1r[t1][1] + xa[2]*w1r[t1][2] + xa[3]*w1r[t1][3]
                  + xb[0]*w1r[t1][4] + xb[1]*w1r[t1][5] + xb[2]*w1r[t1][6] + xb[3]*w1r[t1][7];
        float m1  = mem1[t1][r];
        float rst = (m1 > THRESH) ? THRESH : 0.f;     // reset from previous mem
        m1 = BETA1 * m1 + cur - rst;
        mem1[t1][r] = m1;
        float sp = (m1 > THRESH) ? 1.f : 0.f;
        s_spk1[m][16 * (2 * w + t1) + lo] = (_Float16)sp;
      }
    }
    __syncthreads();   // spk1 ready

    // ---- phase 3: layer-2 GEMM via v_wmma_f32_16x16x32_f16, K=256 in 8 chunks.
    // mem2 update fused into the accumulator chain: C0 = beta2*mem2 - reset.
    // A-fragment LDS loads are software-pipelined one chunk ahead so the
    // ds_load latency is covered by the previous WMMA instead of s_wait_dscnt 0.
    // A layout (16x32 f16): lane(lo,hi): row m=lo, slots0-15 K=32c+16hi+j
    // (matches B) -> two 16B ds loads per chunk.
    v8h a0 = *(const v8h*)&s_spk1[lo][16 * hi];
    v8h a1 = *(const v8h*)&s_spk1[lo][16 * hi + 8];

    v8f c2;
    #pragma unroll
    for (int r = 0; r < 8; ++r) {
      float m2v = mem2[r];
      float rst = (m2v > THRESH) ? THRESH : 0.f;
      c2[r] = BETA2 * m2v - rst;
    }
    #pragma unroll
    for (int c = 0; c < 8; ++c) {
      v8h n0, n1;
      if (c < 7) {   // prefetch next chunk's A fragment before consuming current
        n0 = *(const v8h*)&s_spk1[lo][32 * (c + 1) + 16 * hi];
        n1 = *(const v8h*)&s_spk1[lo][32 * (c + 1) + 16 * hi + 8];
      }
      v16h a = __builtin_shufflevector(a0, a1, 0,1,2,3,4,5,6,7,8,9,10,11,12,13,14,15);
      c2 = __builtin_amdgcn_wmma_f32_16x16x32_f16(false, a, false, b2f[c],
                                                  (short)0, c2, false, false);
      if (c < 7) { a0 = n0; a1 = n1; }
    }

    // ---- state update + streaming NT stores (write-bound: 268MB written once,
    // never re-read -> bypass L2 residency with non-temporal hint)
    const size_t obase = (size_t)t * BATCH * H2 + (size_t)bBase * H2 + 16 * w + lo;
    #pragma unroll
    for (int r = 0; r < 8; ++r) {
      const int m = r + 8 * hi;
      float mm = c2[r];
      float sp = (mm > THRESH) ? 1.f : 0.f;
      mem2[r] = mm;
      spkSum[r] += sp;
      __builtin_nontemporal_store(sp, &spk_out[obase + (size_t)m * H2]);
      __builtin_nontemporal_store(mm, &mem_out[obase + (size_t)m * H2]);
    }
  }

  // ---- head: mean firing -> sigmoid(mean @ Wout^T + b), deterministic reduction
  #pragma unroll
  for (int r = 0; r < 8; ++r) {
    const int m = r + 8 * hi;
    s_mean[m][16 * w + lo] = spkSum[r] * (1.0f / T_STEPS);
  }
  __syncthreads();

  if (tid < BT) {
    float acc = bout[0];
    for (int n = 0; n < H2; ++n) acc += s_mean[tid][n] * Wout[n];
    prob_out[bBase + tid] = 1.f / (1.f + expf(-acc));
  }
}

extern "C" void kernel_launch(void* const* d_in, const int* in_sizes, int n_in,
                              void* d_out, int out_size, void* d_ws, size_t ws_size,
                              hipStream_t stream) {
  const float* x    = (const float*)d_in[0];
  const float* W1   = (const float*)d_in[1];
  const float* W2   = (const float*)d_in[2];
  const float* Wout = (const float*)d_in[3];
  const float* bout = (const float*)d_in[4];
  float* spk  = (float*)d_out;
  float* mem  = spk + (size_t)T_STEPS * BATCH * H2;
  float* prob = mem + (size_t)T_STEPS * BATCH * H2;
  lif_snn_kernel<<<dim3(BATCH / BT), dim3(256), 0, stream>>>(x, W1, W2, Wout, bout,
                                                             spk, mem, prob);
}
